// Attention_49778670961061
// MI455X (gfx1250) — matrix operand order repaired and checked by reading, not compiled, not run on hardware
//
#include <hip/hip_runtime.h>
#include <hip/hip_bf16.h>
#include <cstdint>
#include <cstddef>

// ---------------------------------------------------------------------------
// Types
// ---------------------------------------------------------------------------
typedef __bf16 bf16_t;
typedef __attribute__((ext_vector_type(16))) __bf16 v16bf;
typedef __attribute__((ext_vector_type(8)))  __bf16 v8bf;
typedef __attribute__((ext_vector_type(8)))  float  v8f;
typedef __attribute__((ext_vector_type(4)))  unsigned v4u;
typedef __attribute__((ext_vector_type(8)))  int      v8i;
typedef __attribute__((ext_vector_type(4)))  int      v4i;

#define D_MODEL 1024
#define N_HEAD  16
#define D_HEAD  64
#define BATCH   2
#define SEQLEN  2048
#define MROWS   (BATCH * SEQLEN)   // 4096

__device__ __forceinline__ __bf16 f2bf(float f) {
    unsigned u = __builtin_bit_cast(unsigned, f);
    u += 0x7FFFu + ((u >> 16) & 1u);          // round-to-nearest-even
    unsigned short h = (unsigned short)(u >> 16);
    return __builtin_bit_cast(__bf16, h);
}

// ---------------------------------------------------------------------------
// CDNA5 data movers
// ---------------------------------------------------------------------------
// Async memory->LDS copy, 16 bytes per lane, tracked with ASYNCcnt.
__device__ __forceinline__ void async_load_to_lds_b128(void* lds,
                                                       const void* gsrc) {
    unsigned l = (unsigned)(uintptr_t)lds;          // addr[31:0] == LDS offset
    asm volatile("global_load_async_to_lds_b128 %0, %1, off"
                 :: "v"(l), "v"(gsrc)
                 : "memory");
}
__device__ __forceinline__ void wait_asynccnt0() {
    asm volatile("s_wait_asynccnt 0x0" ::: "memory");
}

// Tensor Data Mover: DMA a 2D tile of bf16 (tile_cols x tile_rows) from
// global (row stride row_stride_elems) into LDS, inserting pad_amount
// (code: 3 -> 4 DWORDs = 16 B) after every row (pad_interval code selects
// the row byte length: 3 -> 64 B, 4 -> 128 B, 5 -> 256 B).
// Descriptor packing per CDNA5 ISA sec. 8.3/8.4; tile == tensor so no OOB.
// This toolchain exposes the 6-arg builtin:
//   (v4u g0, v8i g1, v4i g2, v4i g3, v8i extra, i32 cpol)
__device__ __forceinline__ void tdm_load_2d_bf16(void* lds_dst,
                                                 const void* gsrc,
                                                 unsigned tile_cols,
                                                 unsigned tile_rows,
                                                 unsigned row_stride_elems,
                                                 unsigned pad_interval_code,
                                                 unsigned pad_amount_code) {
    const unsigned lds = (unsigned)(uintptr_t)lds_dst;
    const unsigned long long ga = (unsigned long long)(uintptr_t)gsrc;
    v4u g0;
    g0[0] = 1u;                                        // count=1, user mode
    g0[1] = lds;                                       // lds_addr
    g0[2] = (unsigned)(ga & 0xFFFFFFFFu);              // global_addr[31:0]
    g0[3] = (unsigned)((ga >> 32) & 0x01FFFFFFu)       // global_addr[56:32]
          | 0x80000000u;                               // type=2 ("image")
    v8i g1;
    g1[0] = (int)((1u << 16)                           // data_size = 2 bytes
                | (1u << 20)                           // pad_enable
                | (pad_interval_code << 22)
                | (pad_amount_code << 25));
    g1[1] = (int)(tile_cols << 16);                    // tensor_dim0[15:0]
    g1[2] = (int)((tile_rows & 0xFFFFu) << 16);        // tensor_dim1[15:0]
    g1[3] = (int)((tile_rows >> 16) | (tile_cols << 16)); // dim1 hi|tile_dim0
    g1[4] = (int)(tile_rows & 0xFFFFu);                // tile_dim1 (dim2=0)
    g1[5] = (int)row_stride_elems;                     // tensor_dim0_stride
    g1[6] = 0;
    g1[7] = 0;
    const v4i z4 = {0, 0, 0, 0};
    const v8i z8 = {0, 0, 0, 0, 0, 0, 0, 0};
    __builtin_amdgcn_tensor_load_to_lds(g0, g1, z4, z4, z8, 0);
}

// ---------------------------------------------------------------------------
// f32 -> bf16 conversion
// ---------------------------------------------------------------------------
__global__ void f32_to_bf16_kernel(const float* __restrict__ in,
                                   __bf16* __restrict__ out, int n) {
    int i = blockIdx.x * blockDim.x + threadIdx.x;
    int stride = gridDim.x * blockDim.x;
    for (; i < n; i += stride) out[i] = f2bf(in[i]);
}

// ---------------------------------------------------------------------------
// Tiled bf16 WMMA GEMM: C[M,N] = A[M,K] (row-major) @ B[K,N] (row-major)
// Block tile 128x128, K-step 32, 256 threads = 8 waves (2 M x 4 N),
// each wave: 64x32 = 4x2 tiles of 16x16. Tiles staged by the TDM.
// MODE 0: store bf16 C0.  MODE 1: store f32 C0.
// MODE 2: N==128, cols [0,64) -> bf16 C0 (K), cols [64,128) -> bf16 C1 (V).
// ---------------------------------------------------------------------------
template <int MODE>
__global__ __launch_bounds__(256) void gemm_bf16_wmma(
    const __bf16* __restrict__ A, const __bf16* __restrict__ Bm,
    void* __restrict__ C0, void* __restrict__ C1, int M, int N, int K) {
    constexpr int BM = 128, BN = 128, BK = 32;
    constexpr int LDA = 40;    // 32 + 4-DWORD TDM pad -> 80 B row stride
    constexpr int LDB = 136;   // 128 + 4-DWORD TDM pad -> 272 B row stride
    __shared__ alignas(16) __bf16 As[BM * LDA];
    __shared__ alignas(16) __bf16 Bs[BK * LDB];

    const int tid  = threadIdx.x;
    const int m0g  = blockIdx.y * BM;
    const int n0g  = blockIdx.x * BN;
    const int w    = tid >> 5;
    const int lane = tid & 31;
    const int half = lane >> 4;
    const int l16  = lane & 15;
    const int wm   = (w >> 2) * 64;   // wave M origin in block tile
    const int wn   = (w & 3) * 32;    // wave N origin in block tile

    v8f acc[4][2] = {};

    for (int k0 = 0; k0 < K; k0 += BK) {
        __syncthreads();
        if (tid < 32) {   // wave 0 drives the tensor DMA engine
            // A tile: 128 rows x 32 cols; row = 64 B -> interval code 3
            tdm_load_2d_bf16(As, A + (size_t)m0g * K + k0,
                             /*cols*/ BK, /*rows*/ BM, /*stride*/ K, 3, 3);
            // B tile: 32 rows x 128 cols; row = 256 B -> interval code 5
            tdm_load_2d_bf16(Bs, Bm + (size_t)k0 * N + n0g,
                             /*cols*/ BN, /*rows*/ BK, /*stride*/ N, 5, 3);
            __builtin_amdgcn_s_wait_tensorcnt(0);
        }
        __syncthreads();

        v16bf afrag[4];
        for (int i = 0; i < 4; ++i) {   // A 16x32 fragment: K = 16*half+e
            const __bf16* base = &As[(wm + 16 * i + l16) * LDA];
            for (int e = 0; e < 8; ++e) afrag[i][e]     = base[16 * half + e];
            for (int e = 0; e < 8; ++e) afrag[i][8 + e] = base[16 * half + 8 + e];
        }
        v16bf bfrag[2];
        for (int j = 0; j < 2; ++j) {   // B 32x16 fragment: K = 16*half+e
            for (int e = 0; e < 16; ++e)
                bfrag[j][e] = Bs[(16 * half + e) * LDB + wn + 16 * j + l16];
        }
        for (int i = 0; i < 4; ++i)
            for (int j = 0; j < 2; ++j)
                acc[i][j] = __builtin_amdgcn_wmma_f32_16x16x32_bf16(
                    false, afrag[i], false, bfrag[j], (short)0, acc[i][j],
                    false, false);
    }

    for (int i = 0; i < 4; ++i)
        for (int j = 0; j < 2; ++j)
            for (int r = 0; r < 8; ++r) {
                const size_t m = (size_t)(m0g + wm + 16 * i + r + 8 * half);
                const int    n = n0g + wn + 16 * j + l16;
                const float  v = acc[i][j][r];
                if (MODE == 0) {
                    ((__bf16*)C0)[m * N + n] = f2bf(v);
                } else if (MODE == 1) {
                    ((float*)C0)[m * N + n] = v;
                } else {
                    if (n < D_HEAD) ((__bf16*)C0)[m * D_HEAD + n] = f2bf(v);
                    else            ((__bf16*)C1)[m * D_HEAD + (n - D_HEAD)] = f2bf(v);
                }
            }
}

// ---------------------------------------------------------------------------
// Flash attention core (bf16 WMMA, f32 accumulate, online softmax).
// Grid: (L/128, N_HEAD, B). Block: 256 threads = 8 waves; wave w owns
// q-rows [16w, 16w+16). K/V processed in tiles of 64 keys (TDM-staged);
// Q staged via async global->LDS loads. K/V shared across heads.
// ---------------------------------------------------------------------------
__global__ __launch_bounds__(256) void flash_attn_wmma(
    const __bf16* __restrict__ Q,   // [B*L, 1024] projected q (bf16)
    const __bf16* __restrict__ Km,  // [B*L, 64]
    const __bf16* __restrict__ Vm,  // [B*L, 64]
    __bf16* __restrict__ O) {       // [B*L, 1024] head-interleaved out
    constexpr int LDQ = 72, LDK = 72, LDV = 72, LDP = 72;  // 16B-multiple
    __shared__ alignas(16) __bf16 Qs[128 * LDQ];
    __shared__ alignas(16) __bf16 Ks[64 * LDK];
    __shared__ alignas(16) __bf16 Vs[64 * LDV];
    __shared__ alignas(16) __bf16 Ps[128 * LDP];

    const int tid  = threadIdx.x;
    const int qt   = blockIdx.x;
    const int h    = blockIdx.y;
    const int b    = blockIdx.z;
    const int w    = tid >> 5;
    const int lane = tid & 31;
    const int half = lane >> 4;
    const int l16  = lane & 15;
    const size_t rowbase = (size_t)b * SEQLEN;

    {   // stage Q block 128x64 with async global->LDS (ASYNCcnt path)
        const int row = tid >> 1, col = (tid & 1) * 32;
        const __bf16* src =
            Q + (rowbase + qt * 128 + row) * D_MODEL + h * D_HEAD + col;
        for (int c = 0; c < 32; c += 8)
            async_load_to_lds_b128(&Qs[row * LDQ + col + c], src + c);
        wait_asynccnt0();
    }

    float mrow[8], lrow[8];
    for (int r = 0; r < 8; ++r) { mrow[r] = -1e30f; lrow[r] = 0.f; }
    v8f accO[4] = {};
    const float scale = 0.125f;                      // 1/sqrt(64)
    const float LOG2E = 1.44269504088896340736f;

    for (int kt = 0; kt < SEQLEN / 64; ++kt) {
        __syncthreads();   // protect Ks/Vs reuse from previous iteration
        if (tid < 32) {    // TDM: 64x64 bf16 tiles, row = 128 B -> code 4
            const size_t kb = (rowbase + (size_t)kt * 64) * D_HEAD;
            tdm_load_2d_bf16(Ks, Km + kb, D_HEAD, 64, D_HEAD, 4, 3);
            tdm_load_2d_bf16(Vs, Vm + kb, D_HEAD, 64, D_HEAD, 4, 3);
            __builtin_amdgcn_s_wait_tensorcnt(0);
        }
        __syncthreads();

        // ---- S = Q * K^T (16 q-rows x 64 keys per wave) ----
        v8f S[4] = {};
        for (int kd = 0; kd < D_HEAD; kd += 32) {
            v16bf a;
            const __bf16* ab = &Qs[(16 * w + l16) * LDQ + kd];
            for (int e = 0; e < 8; ++e) a[e]     = ab[16 * half + e];
            for (int e = 0; e < 8; ++e) a[8 + e] = ab[16 * half + 8 + e];
            for (int j = 0; j < 4; ++j) {
                v16bf bfr;   // B[k][n] = K[key=n][d=k]: contiguous row read
                const __bf16* bb = &Ks[(16 * j + l16) * LDK + kd + 16 * half];
                for (int e = 0; e < 16; ++e) bfr[e] = bb[e];
                S[j] = __builtin_amdgcn_wmma_f32_16x16x32_bf16(
                    false, a, false, bfr, (short)0, S[j], false, false);
            }
        }

        // ---- online softmax (rows r+8*half, 16-lane reductions) ----
        float rmax[8];
        for (int r = 0; r < 8; ++r) {
            float mx = -1e30f;
            for (int j = 0; j < 4; ++j) mx = fmaxf(mx, S[j][r] * scale);
            rmax[r] = mx;
        }
        for (int d = 1; d < 16; d <<= 1)
            for (int r = 0; r < 8; ++r)
                rmax[r] = fmaxf(rmax[r], __shfl_xor(rmax[r], d, 32));

        float corr[8], rsum[8];
        for (int r = 0; r < 8; ++r) {
            const float mn = fmaxf(mrow[r], rmax[r]);
            corr[r] = exp2f((mrow[r] - mn) * LOG2E);
            mrow[r] = mn;
            rsum[r] = 0.f;
        }
        for (int j = 0; j < 4; ++j)
            for (int r = 0; r < 8; ++r) {
                const float p = exp2f((S[j][r] * scale - mrow[r]) * LOG2E);
                rsum[r] += p;
                Ps[(16 * w + r + 8 * half) * LDP + 16 * j + l16] = f2bf(p);
            }
        for (int d = 1; d < 16; d <<= 1)
            for (int r = 0; r < 8; ++r) rsum[r] += __shfl_xor(rsum[r], d, 32);
        for (int r = 0; r < 8; ++r) lrow[r] = lrow[r] * corr[r] + rsum[r];
        for (int t = 0; t < 4; ++t)
            for (int r = 0; r < 8; ++r) accO[t][r] *= corr[r];
        __syncthreads();   // Ps write -> read ordering (rows are wave-local)

        // ---- O += P * V  (P: 16x64, V: 64x64) ----
        for (int kd = 0; kd < 64; kd += 32) {
            v16bf a;   // A-fragment of P from LDS round-trip
            const __bf16* ab = &Ps[(16 * w + l16) * LDP + kd];
            for (int e = 0; e < 8; ++e) a[e]     = ab[16 * half + e];
            for (int e = 0; e < 8; ++e) a[8 + e] = ab[16 * half + 8 + e];
            for (int t = 0; t < 4; ++t) {
                v16bf bfr;   // B[k][n] = V[key=k][d=n]
                for (int e = 0; e < 16; ++e)
                    bfr[e] = Vs[(kd + 16 * half + e) * LDV + 16 * t + l16];
                accO[t] = __builtin_amdgcn_wmma_f32_16x16x32_bf16(
                    false, a, false, bfr, (short)0, accO[t], false, false);
            }
        }
    }

    // ---- normalize and store head-interleaved output ----
    for (int t = 0; t < 4; ++t)
        for (int r = 0; r < 8; ++r) {
            const size_t row = rowbase + qt * 128 + 16 * w + r + 8 * half;
            const int    col = h * D_HEAD + 16 * t + l16;
            O[row * D_MODEL + col] = f2bf(accO[t][r] / lrow[r]);
        }
}

// ---------------------------------------------------------------------------
// Host-side orchestration
// ---------------------------------------------------------------------------
extern "C" void kernel_launch(void* const* d_in, const int* in_sizes, int n_in,
                              void* d_out, int out_size, void* d_ws,
                              size_t ws_size, hipStream_t stream) {
    (void)in_sizes; (void)n_in; (void)out_size; (void)ws_size;
    const float* q    = (const float*)d_in[0];
    const float* kv   = (const float*)d_in[1];
    const float* w_q  = (const float*)d_in[2];
    const float* w_kv = (const float*)d_in[3];
    const float* w_c  = (const float*)d_in[4];

    size_t off = 0;
    auto alloc_bf = [&](size_t elems) -> __bf16* {
        __bf16* p = (__bf16*)((char*)d_ws + off);
        off = (off + elems * sizeof(__bf16) + 255) & ~(size_t)255;
        return p;
    };
    __bf16* q_bf   = alloc_bf((size_t)MROWS * D_MODEL);
    __bf16* kv_bf  = alloc_bf((size_t)MROWS * D_MODEL);
    __bf16* wq_bf  = alloc_bf((size_t)D_MODEL * D_MODEL);
    __bf16* wkv_bf = alloc_bf((size_t)D_MODEL * 2 * D_HEAD);
    __bf16* wc_bf  = alloc_bf((size_t)D_MODEL * D_MODEL);
    __bf16* qp_bf  = alloc_bf((size_t)MROWS * D_MODEL);
    __bf16* k_bf   = alloc_bf((size_t)MROWS * D_HEAD);
    __bf16* v_bf   = alloc_bf((size_t)MROWS * D_HEAD);
    __bf16* ao_bf  = alloc_bf((size_t)MROWS * D_MODEL);

    const int CT = 256;
    f32_to_bf16_kernel<<<1024, CT, 0, stream>>>(q,    q_bf,   MROWS * D_MODEL);
    f32_to_bf16_kernel<<<1024, CT, 0, stream>>>(kv,   kv_bf,  MROWS * D_MODEL);
    f32_to_bf16_kernel<<<1024, CT, 0, stream>>>(w_q,  wq_bf,  D_MODEL * D_MODEL);
    f32_to_bf16_kernel<<<256,  CT, 0, stream>>>(w_kv, wkv_bf, D_MODEL * 2 * D_HEAD);
    f32_to_bf16_kernel<<<1024, CT, 0, stream>>>(w_c,  wc_bf,  D_MODEL * D_MODEL);

    // qp = q @ w_q : (4096 x 1024) @ (1024 x 1024) -> bf16
    gemm_bf16_wmma<0><<<dim3(D_MODEL / 128, MROWS / 128), 256, 0, stream>>>(
        q_bf, wq_bf, qp_bf, nullptr, MROWS, D_MODEL, D_MODEL);
    // kvp = kv @ w_kv : (4096 x 1024) @ (1024 x 128) -> split K / V (bf16)
    gemm_bf16_wmma<2><<<dim3(1, MROWS / 128), 256, 0, stream>>>(
        kv_bf, wkv_bf, k_bf, v_bf, MROWS, 2 * D_HEAD, D_MODEL);
    // attention core (K/V shared across heads)
    flash_attn_wmma<<<dim3(SEQLEN / 128, N_HEAD, BATCH), 256, 0, stream>>>(
        qp_bf, k_bf, v_bf, ao_bf);
    // out = attn_out @ w_concat : (4096 x 1024) @ (1024 x 1024) -> f32
    gemm_bf16_wmma<1><<<dim3(D_MODEL / 128, MROWS / 128), 256, 0, stream>>>(
        ao_bf, wc_bf, (float*)d_out, nullptr, MROWS, D_MODEL, D_MODEL);
}
